// GNNPolicy_cl_9792525435393
// MI455X (gfx1250) — matrix operand order repaired and checked by reading, not compiled, not run on hardware
//
#include <hip/hip_runtime.h>

// ---------------------------------------------------------------------------
// GNN policy forward for MI455X (gfx1250, wave32, WMMA).
// All dense K>=64 GEMMs run through v_wmma_f32_16x16x32_f16 (f16 in, f32 acc).
// Edge-level segment softmax uses f32 global atomics (bandwidth-bound path).
// ---------------------------------------------------------------------------

typedef __attribute__((ext_vector_type(16))) _Float16 v16h;
typedef __attribute__((ext_vector_type(8)))  float    v8f;

#define LEAKY 0.2f

__device__ inline v8f v8f_zero() {
  v8f z;
#pragma unroll
  for (int i = 0; i < 8; ++i) z[i] = 0.0f;
  return z;
}

// B fragment: lane holds column n = n0 + (lane&15); half-wave selects K group.
// 16 contiguous f32 from W row n, converted to f16.
__device__ inline v16h frag_b_f32(const float* __restrict__ p) {
  v16h r;
#pragma unroll
  for (int i = 0; i < 16; ++i) r[i] = (_Float16)p[i];
  return r;
}

// A fragment: lanes 0-15 hold K {0..15}, lanes 16-31 hold {16..31} (contiguous, as B).
__device__ inline v16h frag_a_f32(const float* __restrict__ plo,
                                  const float* __restrict__ phi) {
  v16h r;
#pragma unroll
  for (int i = 0; i < 8; ++i) r[i] = (_Float16)plo[i];
#pragma unroll
  for (int i = 0; i < 8; ++i) r[8 + i] = (_Float16)phi[i];
  return r;
}

// C = act(concat[A0|A1] @ W^T + bias), N = 64 fixed, K = K0+K1 compile-time.
// One wave computes a 32(M) x 64(N) tile; 128-thread block = 4 waves = 128 rows.
template <int K0, int K1>
__global__ void __launch_bounds__(128)
gemm_n64(const float* __restrict__ A0,
         const float* __restrict__ A1,
         const float* __restrict__ W,
         const float* __restrict__ bias,
         float* __restrict__ C, int M, int act) {
  constexpr int K = K0 + K1;
  const int lane = threadIdx.x & 31;
  const int wave = threadIdx.x >> 5;
  const int tileRow = (blockIdx.x * 4 + wave) * 32;   // wave-uniform
  if (tileRow >= M) return;                            // whole wave exits; EXEC stays full
  const int hs  = lane >> 4;
  const int l16 = lane & 15;

  v8f acc[2][4];
#pragma unroll
  for (int mt = 0; mt < 2; ++mt)
#pragma unroll
    for (int nt = 0; nt < 4; ++nt) acc[mt][nt] = v8f_zero();

#pragma unroll
  for (int kc = 0; kc < (K >> 5); ++kc) {
    constexpr int dummy = 0; (void)dummy;
    const int kbase = kc * 32;

    v16h b[4];
#pragma unroll
    for (int nt = 0; nt < 4; ++nt) {
      const int n = nt * 16 + l16;
      b[nt] = frag_b_f32(W + (size_t)n * K + kbase + hs * 16);
    }

    v16h a[2];
#pragma unroll
    for (int mt = 0; mt < 2; ++mt) {
      int row = tileRow + mt * 16 + l16;
      if (row >= M) row = M - 1;                       // clamp keeps loads in-bounds
      const float* rp;
      if (kbase < K0) rp = A0 + (size_t)row * K0 + kbase;
      else            rp = A1 + (size_t)row * K1 + (kbase - K0);
      a[mt] = frag_a_f32(rp + hs * 16, rp + hs * 16 + 8);
    }

#pragma unroll
    for (int mt = 0; mt < 2; ++mt)
#pragma unroll
      for (int nt = 0; nt < 4; ++nt)
        acc[mt][nt] = __builtin_amdgcn_wmma_f32_16x16x32_f16(
            false, a[mt], false, b[nt], (short)0, acc[mt][nt], false, false);
  }

  // Hoisted bias (uniform branch, 4 loads)
  float bv[4];
#pragma unroll
  for (int nt = 0; nt < 4; ++nt) bv[nt] = bias ? bias[nt * 16 + l16] : 0.0f;

  if (tileRow + 32 <= M) {
    // Fast path: full tile, branch-free stores (always taken when M % 32 == 0).
#pragma unroll
    for (int mt = 0; mt < 2; ++mt)
#pragma unroll
      for (int nt = 0; nt < 4; ++nt) {
        const int col = nt * 16 + l16;
#pragma unroll
        for (int r = 0; r < 8; ++r) {
          const int row = tileRow + mt * 16 + hs * 8 + r;
          float v = acc[mt][nt][r] + bv[nt];
          if (act) v = fmaxf(v, 0.0f);
          C[(size_t)row * 64 + col] = v;
        }
      }
  } else {
#pragma unroll
    for (int mt = 0; mt < 2; ++mt)
#pragma unroll
      for (int nt = 0; nt < 4; ++nt) {
        const int col = nt * 16 + l16;
#pragma unroll
        for (int r = 0; r < 8; ++r) {
          const int row = tileRow + mt * 16 + hs * 8 + r;
          if (row < M) {
            float v = acc[mt][nt][r] + bv[nt];
            if (act) v = fmaxf(v, 0.0f);
            C[(size_t)row * 64 + col] = v;
          }
        }
      }
  }
}

// First-layer MLPs with tiny K (5 / 19 / 1): VALU, one thread per (node, out-chan).
__global__ void embed_small(const float* __restrict__ X, int M, int F,
                            const float* __restrict__ shift,
                            const float* __restrict__ scale,
                            const float* __restrict__ W,   // [64, F]
                            const float* __restrict__ b,   // [64]
                            float* __restrict__ Y, int act) {
  long long t = (long long)blockIdx.x * blockDim.x + threadIdx.x;
  if (t >= (long long)M * 64) return;
  const int n = (int)(t >> 6), k = (int)(t & 63);
  float acc = b[k];
  for (int f = 0; f < F; ++f)
    acc += W[k * F + f] * ((X[(size_t)n * F + f] + shift[f]) * scale[f]);
  if (act) acc = fmaxf(acc, 0.0f);
  Y[t] = acc;
}

__global__ void init_softmax(float* __restrict__ mx, float* __restrict__ den, int n) {
  int i = blockIdx.x * blockDim.x + threadIdx.x;
  if (i < n) { mx[i] = -__builtin_inff(); den[i] = 0.0f; }
}

__global__ void zero_f32(float* __restrict__ p, long long n) {
  long long i = (long long)blockIdx.x * blockDim.x + threadIdx.x;
  if (i < n) p[i] = 0.0f;
}

// Pass A: logit = leaky_relu(hl[s]+he[e]+hr[d]) . att ; segment max via atomics.
__global__ void edge_logit_max(const int* __restrict__ src, const int* __restrict__ dst,
                               const float* __restrict__ hl, const float* __restrict__ he,
                               const float* __restrict__ hr, const float* __restrict__ att,
                               float* __restrict__ logit, float* __restrict__ mx, int E) {
  int e = blockIdx.x * blockDim.x + threadIdx.x;
  if (e >= E) return;
  const int s = src[e], d = dst[e];
  const float4* pl = (const float4*)(hl + (size_t)s * 64);
  const float4* pe = (const float4*)(he + (size_t)e * 64);
  const float4* pr = (const float4*)(hr + (size_t)d * 64);
  const float4* pa = (const float4*)att;
  float acc = 0.0f;
#pragma unroll
  for (int i = 0; i < 16; ++i) {
    const float4 a = pl[i], b = pe[i], c = pr[i], w = pa[i];
    float h;
    h = a.x + b.x + c.x; h = h > 0.0f ? h : LEAKY * h; acc += h * w.x;
    h = a.y + b.y + c.y; h = h > 0.0f ? h : LEAKY * h; acc += h * w.y;
    h = a.z + b.z + c.z; h = h > 0.0f ? h : LEAKY * h; acc += h * w.z;
    h = a.w + b.w + c.w; h = h > 0.0f ? h : LEAKY * h; acc += h * w.w;
  }
  logit[e] = acc;
  // float atomic max via signed/unsigned int monotonic trick
  if (acc >= 0.0f) atomicMax((int*)(mx + d), __float_as_int(acc));
  else             atomicMin((unsigned int*)(mx + d), (unsigned int)__float_as_int(acc));
}

// Pass B: ex = exp(logit - mx[d]); den[d] += ex.
__global__ void edge_exp_sum(const int* __restrict__ dst, const float* __restrict__ logit,
                             const float* __restrict__ mx, float* __restrict__ ex,
                             float* __restrict__ den, int E) {
  int e = blockIdx.x * blockDim.x + threadIdx.x;
  if (e >= E) return;
  const int d = dst[e];
  const float x = __expf(logit[e] - mx[d]);   // mx finite for any d with edges
  ex[e] = x;
  atomicAdd(den + d, x);
}

// Pass C: agg[d] += post_scale * alpha * hj (recompute hj; 64 f32 atomics/edge).
__global__ void edge_aggregate(const int* __restrict__ src, const int* __restrict__ dst,
                               const float* __restrict__ hl, const float* __restrict__ he,
                               const float* __restrict__ hr, const float* __restrict__ ex,
                               const float* __restrict__ den,
                               const float* __restrict__ post_scale,
                               float* __restrict__ agg, int E) {
  int e = blockIdx.x * blockDim.x + threadIdx.x;
  if (e >= E) return;
  const int s = src[e], d = dst[e];
  const float a = post_scale[0] * (ex[e] / (den[d] + 1e-8f));
  const float4* pl = (const float4*)(hl + (size_t)s * 64);
  const float4* pe = (const float4*)(he + (size_t)e * 64);
  const float4* pr = (const float4*)(hr + (size_t)d * 64);
  float* out = agg + (size_t)d * 64;
#pragma unroll
  for (int i = 0; i < 16; ++i) {
    const float4 A = pl[i], B = pe[i], C = pr[i];
    atomicAdd(out + i * 4 + 0, a * (A.x + B.x + C.x));
    atomicAdd(out + i * 4 + 1, a * (A.y + B.y + C.y));
    atomicAdd(out + i * 4 + 2, a * (A.z + B.z + C.z));
    atomicAdd(out + i * 4 + 3, a * (A.w + B.w + C.w));
  }
}

// Final 64 -> 1 head.
__global__ void head_final(const float* __restrict__ H, const float* __restrict__ w,
                           float* __restrict__ out, int M) {
  int n = blockIdx.x * blockDim.x + threadIdx.x;
  if (n >= M) return;
  const float4* ph = (const float4*)(H + (size_t)n * 64);
  const float4* pw = (const float4*)w;
  float acc = 0.0f;
#pragma unroll
  for (int i = 0; i < 16; ++i) {
    const float4 h = ph[i], ww = pw[i];
    acc += h.x * ww.x + h.y * ww.y + h.z * ww.z + h.w * ww.w;
  }
  out[n] = acc;
}

// ---------------------------------------------------------------------------
static inline void gemm64(hipStream_t st, const float* A, const float* W,
                          const float* bias, float* C, int M, int act) {
  gemm_n64<64, 0><<<(M + 127) / 128, 128, 0, st>>>(A, nullptr, W, bias, C, M, act);
}
static inline void gemm128(hipStream_t st, const float* A0, const float* A1,
                           const float* W, const float* bias, float* C, int M, int act) {
  gemm_n64<64, 64><<<(M + 127) / 128, 128, 0, st>>>(A0, A1, W, bias, C, M, act);
}

static void run_gat(hipStream_t st,
                    const float* left, int nLeft, float* right /*in: [nDst,64]*/,
                    int nDst, const int* src, const int* dst,
                    const float* e_emb, int E,
                    const float* Wl, const float* bl, const float* We, const float* Wr,
                    const float* att, const float* post_scale,
                    const float* Wo1, const float* bo1, const float* Wo2, const float* bo2,
                    float* he, float* hl, float* hr, float* agg, float* h1,
                    float* logit, float* ex, float* mx, float* den) {
  const int eb = (E + 255) / 256;
  // Node / edge projections (WMMA GEMMs)
  gemm64(st, left,  Wl, bl,      hl, nLeft, 0);
  gemm64(st, right, Wr, nullptr, hr, nDst,  0);
  gemm64(st, e_emb, We, nullptr, he, E,     0);
  // Segment softmax state
  init_softmax<<<(nDst + 255) / 256, 256, 0, st>>>(mx, den, nDst);
  zero_f32<<<(int)(((long long)nDst * 64 + 255) / 256), 256, 0, st>>>(agg, (long long)nDst * 64);
  // Three edge passes
  edge_logit_max<<<eb, 256, 0, st>>>(src, dst, hl, he, hr, att, logit, mx, E);
  edge_exp_sum<<<eb, 256, 0, st>>>(dst, logit, mx, ex, den, E);
  edge_aggregate<<<eb, 256, 0, st>>>(src, dst, hl, he, hr, ex, den, post_scale, agg, E);
  // Output MLP: relu(concat[agg, right] @ Wo1^T + bo1) @ Wo2^T + bo2 -> right
  gemm128(st, agg, right, Wo1, bo1, h1, nDst, 1);
  gemm64(st, h1, Wo2, bo2, right, nDst, 0);   // overwrite right buffer
}

extern "C" void kernel_launch(void* const* d_in, const int* in_sizes, int n_in,
                              void* d_out, int out_size, void* d_ws, size_t ws_size,
                              hipStream_t stream) {
  // Input order: setup_inputs() insertion order, params nested in insertion order.
  const float* cons = (const float*)d_in[0];
  const int*   ei   = (const int*)d_in[1];
  const float* ef   = (const float*)d_in[2];
  const float* var  = (const float*)d_in[3];
  // d_in[4] = variable_features_batch (unused by reference math)
  const float* cons_shift = (const float*)d_in[5];
  const float* cons_scale = (const float*)d_in[6];
  const float* cW1 = (const float*)d_in[7];
  const float* cb1 = (const float*)d_in[8];
  const float* cW2 = (const float*)d_in[9];
  const float* cb2 = (const float*)d_in[10];
  const float* e_shift = (const float*)d_in[11];
  const float* e_scale = (const float*)d_in[12];
  const float* eW = (const float*)d_in[13];
  const float* ebv = (const float*)d_in[14];
  const float* v_shift = (const float*)d_in[15];
  const float* v_scale = (const float*)d_in[16];
  const float* vW1 = (const float*)d_in[17];
  const float* vb1 = (const float*)d_in[18];
  const float* vW2 = (const float*)d_in[19];
  const float* vb2 = (const float*)d_in[20];
  const float* const* g1 = (const float* const*)(d_in + 21);  // Wl bl We Wr att ps Wo1 bo1 Wo2 bo2
  const float* const* g2 = (const float* const*)(d_in + 31);
  const float* oW1 = (const float*)d_in[41];
  const float* ob1 = (const float*)d_in[42];
  const float* oW2 = (const float*)d_in[43];

  const int NC = in_sizes[0] / 5;
  const int NV = in_sizes[3] / 19;
  const int E  = in_sizes[2];
  const int NMAX = NC > NV ? NC : NV;

  // Workspace layout (f32)
  float* W = (float*)d_ws;
  size_t off = 0;
  auto alloc = [&](size_t n) { float* p = W + off; off += n; return p; };
  float* e_emb = alloc((size_t)E * 64);
  float* he    = alloc((size_t)E * 64);
  float* c_emb = alloc((size_t)NC * 64);
  float* v_emb = alloc((size_t)NV * 64);
  float* hl    = alloc((size_t)NMAX * 64);
  float* hr    = alloc((size_t)NMAX * 64);
  float* agg   = alloc((size_t)NMAX * 64);
  float* h1    = alloc((size_t)NMAX * 64);
  float* logit = alloc((size_t)E);
  float* exb   = alloc((size_t)E);
  float* mx    = alloc((size_t)NMAX);
  float* den   = alloc((size_t)NMAX);
  (void)ws_size; (void)n_in; (void)out_size;

  // Embeddings: tiny-K first layers in VALU, 64x64 second layers via WMMA.
  embed_small<<<(int)(((long long)NC * 64 + 255) / 256), 256, 0, stream>>>(
      cons, NC, 5, cons_shift, cons_scale, cW1, cb1, hl, 1);
  gemm64(stream, hl, cW2, cb2, c_emb, NC, 1);
  embed_small<<<(int)(((long long)NV * 64 + 255) / 256), 256, 0, stream>>>(
      var, NV, 19, v_shift, v_scale, vW1, vb1, hl, 1);
  gemm64(stream, hl, vW2, vb2, v_emb, NV, 1);
  embed_small<<<(int)(((long long)E * 64 + 255) / 256), 256, 0, stream>>>(
      ef, E, 1, e_shift, e_scale, eW, ebv, e_emb, 0);

  // GAT 1: variables -> constraints (src = ei[1], dst = ei[0]); updates c_emb.
  run_gat(stream, v_emb, NV, c_emb, NC, ei + E, ei, e_emb, E,
          g1[0], g1[1], g1[2], g1[3], g1[4], g1[5], g1[6], g1[7], g1[8], g1[9],
          he, hl, hr, agg, h1, logit, exb, mx, den);
  // GAT 2: constraints -> variables (src = ei[0], dst = ei[1]); updates v_emb.
  run_gat(stream, c_emb, NC, v_emb, NV, ei, ei + E, e_emb, E,
          g2[0], g2[1], g2[2], g2[3], g2[4], g2[5], g2[6], g2[7], g2[8], g2[9],
          he, hl, hr, agg, h1, logit, exb, mx, den);

  // Output head: relu(v @ oW1^T + ob1) @ oW2^T -> [NV]
  gemm64(stream, v_emb, oW1, ob1, h1, NV, 1);
  head_final<<<(NV + 255) / 256, 256, 0, stream>>>(h1, oW2, (float*)d_out, NV);
}
